// VGAE_14955076125211
// MI455X (gfx1250) — matrix operand order repaired and checked by reading, not compiled, not run on hardware
//
#include <hip/hip_runtime.h>
#include <hip/hip_bf16.h>

typedef __attribute__((ext_vector_type(16))) _Float16 v16h;
typedef __attribute__((ext_vector_type(8)))  float    v8f;

#define NN   8192
#define EE   131072
#define FF   512
#define H1   32
#define H2   16
#define POSW 10.0f

// ---- workspace layout (float offsets) ----
// zero-initialized region first (atomically accumulated buffers)
#define OFF_ACC   0                       // [0]=bce sum, [1]=kld sum (16 floats pad)
#define OFF_DEG   16                      // N
#define OFF_AGG1  (OFF_DEG + NN)          // N*H1
#define OFF_AGGMU (OFF_AGG1 + NN*H1)      // N*H2
#define OFF_AGGLV (OFF_AGGMU + NN*H2)     // N*H2
#define ZERO_END  (OFF_AGGLV + NN*H2)
// non-zeroed scratch
#define OFF_DINV  ZERO_END                // N
#define OFF_H     (OFF_DINV + NN)         // N*H1
#define OFF_H1    (OFF_H + NN*H1)         // N*H1
#define OFF_T2    (OFF_H1 + NN*H1)        // N*H2
#define OFF_T3    (OFF_T2 + NN*H2)        // N*H2
#define OFF_Z16F  (OFF_T3 + NN*H2)        // start of f16 z (N*32 halves), 16B aligned

// ---------------- init: zero atomic accumulation region ----------------
__global__ void k_zero(float* ws) {
    int i = blockIdx.x * blockDim.x + threadIdx.x;
    if (i < ZERO_END) ws[i] = 0.0f;
}

// ---------------- degree scatter: deg[dst] += 1 ----------------
__global__ void k_deg(const int* dst, float* deg) {
    int e = blockIdx.x * blockDim.x + threadIdx.x;
    if (e < EE) atomicAdd(&deg[dst[e]], 1.0f);
}

// ---------------- dinv = (deg + 1)^-1/2  (self loop) ----------------
__global__ void k_dinv(const float* deg, float* dinv) {
    int i = blockIdx.x * blockDim.x + threadIdx.x;
    if (i < NN) dinv[i] = rsqrtf(deg[i] + 1.0f);
}

// ---------------- h = x @ W1  ([N,512] x [512,32]) ----------------
__global__ void k_gemm1(const float* __restrict__ x, const float* __restrict__ W1,
                        float* __restrict__ h) {
    __shared__ float sX[8 * FF];              // 16 KB: 8 rows of x
    int r0 = blockIdx.x * 8;
    for (int idx = threadIdx.x; idx < 8 * FF; idx += 256)
        sX[idx] = x[(size_t)r0 * FF + idx];
    __syncthreads();
    int r = threadIdx.x >> 5;                  // 0..7
    int c = threadIdx.x & 31;                  // 0..31
    float acc = 0.0f;
    #pragma unroll 8
    for (int k = 0; k < FF; ++k)
        acc = fmaf(sX[r * FF + k], W1[k * H1 + c], acc);
    h[(size_t)(r0 + r) * H1 + c] = acc;
}

// ---------------- edge aggregation layer 1 ----------------
__global__ void k_agg1(const int* __restrict__ src, const int* __restrict__ dst,
                       const float* __restrict__ dinv, const float* __restrict__ h,
                       float* __restrict__ agg) {
    int t = blockIdx.x * blockDim.x + threadIdx.x;   // E*32 threads
    int e = t >> 5, c = t & 31;
    if (e >= EE) return;
    int s = src[e], d = dst[e];
    float coef = dinv[s] * dinv[d];
    atomicAdd(&agg[(size_t)d * H1 + c], h[(size_t)s * H1 + c] * coef);
}

// ---------------- h1 = relu(agg + h * dinv^2) ----------------
__global__ void k_h1(const float* __restrict__ agg, const float* __restrict__ h,
                     const float* __restrict__ dinv, float* __restrict__ h1) {
    int t = blockIdx.x * blockDim.x + threadIdx.x;   // N*32
    int i = t >> 5;
    float di = dinv[i];
    float v = agg[t] + h[t] * di * di;
    h1[t] = fmaxf(v, 0.0f);
}

// ---------------- t2 = h1 @ W2 ; t3 = h1 @ W3 (K=32) ----------------
__global__ void k_gemm23(const float* __restrict__ h1, const float* __restrict__ W2,
                         const float* __restrict__ W3, float* __restrict__ t2,
                         float* __restrict__ t3) {
    int t = blockIdx.x * blockDim.x + threadIdx.x;   // N*32 threads
    int row = t >> 5, cc = t & 31;
    const float* W = (cc < H2) ? W2 : W3;
    int c = cc & 15;
    float acc = 0.0f;
    #pragma unroll
    for (int k = 0; k < H1; ++k)
        acc = fmaf(h1[(size_t)row * H1 + k], W[k * H2 + c], acc);
    if (cc < H2) t2[(size_t)row * H2 + c] = acc;
    else         t3[(size_t)row * H2 + c] = acc;
}

// ---------------- edge aggregation layer 2 (mu and logvar) ----------------
__global__ void k_agg2(const int* __restrict__ src, const int* __restrict__ dst,
                       const float* __restrict__ dinv, const float* __restrict__ t2,
                       const float* __restrict__ t3, float* __restrict__ aggmu,
                       float* __restrict__ agglv) {
    int t = blockIdx.x * blockDim.x + threadIdx.x;   // E*32 threads
    int e = t >> 5, cc = t & 31;
    if (e >= EE) return;
    int s = src[e], d = dst[e];
    float coef = dinv[s] * dinv[d];
    int c = cc & 15;
    if (cc < H2) atomicAdd(&aggmu[(size_t)d * H2 + c], t2[(size_t)s * H2 + c] * coef);
    else         atomicAdd(&agglv[(size_t)d * H2 + c], t3[(size_t)s * H2 + c] * coef);
}

// ---- finalize: mu, logvar, z = eps*exp(lv)+mu (f16 padded), KLD partial ----
__global__ void k_final(const float* __restrict__ aggmu, const float* __restrict__ agglv,
                        const float* __restrict__ t2, const float* __restrict__ t3,
                        const float* __restrict__ dinv, const float* __restrict__ eps,
                        float* __restrict__ mu_out, _Float16* __restrict__ z16,
                        float* __restrict__ acc) {
    __shared__ float sred[256];
    int t = blockIdx.x * blockDim.x + threadIdx.x;   // N*16
    int i = t >> 4, j = t & 15;
    float di = dinv[i], d2 = di * di;
    float mu = aggmu[t] + t2[t] * d2;
    float lv = agglv[t] + t3[t] * d2;
    mu_out[t] = mu;                                   // d_out[1 + t]
    float elv = __expf(lv);
    float z = eps[t] * elv + mu;
    z16[(size_t)i * 32 + j]      = (_Float16)z;
    z16[(size_t)i * 32 + 16 + j] = (_Float16)0.0f;    // K padding for WMMA
    float kterm = 1.0f + 2.0f * lv - mu * mu - elv * elv;
    // block reduce
    sred[threadIdx.x] = kterm;
    __syncthreads();
    for (int st = 128; st > 0; st >>= 1) {
        if (threadIdx.x < st) sred[threadIdx.x] += sred[threadIdx.x + st];
        __syncthreads();
    }
    if (threadIdx.x == 0) atomicAdd(&acc[1], sred[0]);
}

// ---- fused decode + BCE: logits tile via WMMA, softplus, reduce ----
// adj_label (256 MB, single-pass stream > L2) is loaded non-temporal so the
// resident z16 tiles (512 KB, re-read by every block) stay hot in L2.
__global__ void __launch_bounds__(256)
k_decode_bce(const _Float16* __restrict__ z16, const int* __restrict__ adj,
             float* __restrict__ acc) {
    __shared__ _Float16 sZi[128 * 32];   // 8 KB
    __shared__ _Float16 sZj[128 * 32];   // 8 KB
    __shared__ float    sred[256];

    int ibase = blockIdx.y * 128;
    int jbase = blockIdx.x * 128;

    const unsigned* gZi = (const unsigned*)(z16 + (size_t)ibase * 32);
    const unsigned* gZj = (const unsigned*)(z16 + (size_t)jbase * 32);
    unsigned* lZi = (unsigned*)sZi;
    unsigned* lZj = (unsigned*)sZj;
    for (int idx = threadIdx.x; idx < 128 * 32 / 2; idx += 256) {
        lZi[idx] = gZi[idx];
        lZj[idx] = gZj[idx];
    }
    __syncthreads();

    int wave = threadIdx.x >> 5;
    int lane = threadIdx.x & 31;
    int ln16 = lane & 15;
    bool lo  = lane < 16;

    float lsum = 0.0f;

    #pragma unroll
    for (int s = 0; s < 8; ++s) {
        int st = wave * 8 + s;           // uniform within wave -> EXEC all-1s
        int ti = st >> 3, tj = st & 7;

        // A: 16x32 f16, M=lane%16; lanes<16 hold K0-15, lanes>=16 K16-31
        int ar = ti * 16 + ln16;
        int kb = lo ? 0 : 16;
        v16h a;
        #pragma unroll
        for (int h = 0; h < 16; ++h) {
            int k = kb + h;
            a[h] = sZi[ar * 32 + k];
        }
        // B: 32x16 f16, col=lane%16; lanes<16 hold K0-15, lanes>=16 K16-31 (zeros)
        int bc = tj * 16 + ln16;
        int ko = lo ? 0 : 16;
        v16h b;
        #pragma unroll
        for (int h = 0; h < 16; ++h)
            b[h] = sZj[bc * 32 + ko + h];

        v8f c = {};
        c = __builtin_amdgcn_wmma_f32_16x16x32_f16(
                false, a, false, b, (short)0, c, false, false);

        // C/D layout: lane<16 -> M=v, lane>=16 -> M=v+8; N=lane%16
        int n = jbase + tj * 16 + ln16;
        int mb = ibase + ti * 16 + (lo ? 0 : 8);
        #pragma unroll
        for (int v = 0; v < 8; ++v) {
            float l = c[v];
            int   y = __builtin_nontemporal_load(&adj[(size_t)(mb + v) * NN + n]);
            float yf = (float)y;
            float e  = __expf(-fabsf(l));
            float sp = __logf(1.0f + e) + fmaxf(-l, 0.0f);   // softplus(-l), stable
            lsum += POSW * yf * sp + (1.0f - yf) * (l + sp);
        }
    }

    sred[threadIdx.x] = lsum;
    __syncthreads();
    for (int stp = 128; stp > 0; stp >>= 1) {
        if (threadIdx.x < stp) sred[threadIdx.x] += sred[threadIdx.x + stp];
        __syncthreads();
    }
    if (threadIdx.x == 0) atomicAdd(&acc[0], sred[0]);
}

// ---------------- scalar epilogue ----------------
__global__ void k_scalar(const float* __restrict__ acc, const float* __restrict__ norm,
                         float* __restrict__ out) {
    float bce = acc[0] / ((float)NN * (float)NN);
    float kld = (-0.5f / (float)NN) * (acc[1] / (float)NN);
    out[0] = norm[0] * bce + kld;
}

extern "C" void kernel_launch(void* const* d_in, const int* in_sizes, int n_in,
                              void* d_out, int out_size, void* d_ws, size_t ws_size,
                              hipStream_t stream) {
    const float* x    = (const float*)d_in[0];
    const int*   eidx = (const int*)d_in[1];
    const int*   adj  = (const int*)d_in[2];
    const float* eps  = (const float*)d_in[3];
    const float* norm = (const float*)d_in[4];
    const float* W1   = (const float*)d_in[5];
    const float* W2   = (const float*)d_in[6];
    const float* W3   = (const float*)d_in[7];
    const int* src = eidx;
    const int* dst = eidx + EE;

    float* ws   = (float*)d_ws;
    float* acc  = ws + OFF_ACC;
    float* deg  = ws + OFF_DEG;
    float* agg1 = ws + OFF_AGG1;
    float* aggm = ws + OFF_AGGMU;
    float* aggl = ws + OFF_AGGLV;
    float* dinv = ws + OFF_DINV;
    float* h    = ws + OFF_H;
    float* h1   = ws + OFF_H1;
    float* t2   = ws + OFF_T2;
    float* t3   = ws + OFF_T3;
    _Float16* z16 = (_Float16*)(ws + OFF_Z16F);

    float* out    = (float*)d_out;
    float* mu_out = out + 1;

    k_zero  <<<(ZERO_END + 255) / 256, 256, 0, stream>>>(ws);
    k_deg   <<<EE / 256, 256, 0, stream>>>(dst, deg);
    k_dinv  <<<NN / 256, 256, 0, stream>>>(deg, dinv);
    k_gemm1 <<<NN / 8, 256, 0, stream>>>(x, W1, h);
    k_agg1  <<<(EE * H1) / 256, 256, 0, stream>>>(src, dst, dinv, h, agg1);
    k_h1    <<<(NN * H1) / 256, 256, 0, stream>>>(agg1, h, dinv, h1);
    k_gemm23<<<(NN * 32) / 256, 256, 0, stream>>>(h1, W2, W3, t2, t3);
    k_agg2  <<<(EE * 32) / 256, 256, 0, stream>>>(src, dst, dinv, t2, t3, aggm, aggl);
    k_final <<<(NN * H2) / 256, 256, 0, stream>>>(aggm, aggl, t2, t3, dinv, eps,
                                                  mu_out, z16, acc);
    dim3 grid(NN / 128, NN / 128);
    k_decode_bce<<<grid, 256, 0, stream>>>(z16, adj, acc);
    k_scalar<<<1, 1, 0, stream>>>(acc, norm, out);
}
